// VortexAttention_35880156791346
// MI455X (gfx1250) — matrix operand order repaired and checked by reading, not compiled, not run on hardware
//
#include <hip/hip_runtime.h>
#include <hip/hip_bf16.h>
#include <math.h>

// ---------------------------------------------------------------------------
// VortexAttention on MI455X (gfx1250):
//   QKV projections as bf16 WMMA (v_wmma_f32_16x16x32_bf16), f32 accumulate.
//   Weights pre-converted once into WMMA B-fragment order in d_ws (1.5 MB).
//   x staged in LDS in A-fragment order; softmax/attention/gating in VALU f32.
// ---------------------------------------------------------------------------

#define NUM_NODES 9
#define HIDDEN    512
#define TB        4                    // batch elements per workgroup
#define MROWS     (TB * NUM_NODES)     // 36 live rows
#define MROWS_PAD 48                   // padded to 3 x 16
#define MTILES    3
#define KCHUNKS   (HIDDEN / 32)        // 16
#define NTILES    (HIDDEN / 16)        // 32
#define THREADS   256
#define WAVES     8

typedef __attribute__((ext_vector_type(16))) __bf16 bf16x16;
typedef __attribute__((ext_vector_type(8)))  float  f32x8;

// LDS layout (bytes)
#define XFRAG_OFF  0
#define XFRAG_SZ   (MTILES * KCHUNKS * 1024)          // 49152
#define QBUF_OFF   (XFRAG_OFF + XFRAG_SZ)             // 49152
#define QKV_SZ     (MROWS * HIDDEN * 2)               // 36864
#define KBUF_OFF   (QBUF_OFF + QKV_SZ)                // 86016
#define VBUF_OFF   (KBUF_OFF + QKV_SZ)                // 122880
#define GATES_OFF  (VBUF_OFF + QKV_SZ)                // 159744
#define SMEM_BYTES (GATES_OFF + TB * 4)               // 159760 (< 160 KB)

// vortex mask (precomputed from the reference graph), rows = query node i
__constant__ float c_mask[81] = {
    1.f,2.f,1.f,1.f,2.f,1.f,1.f,2.f,1.5f,   // 0
    2.f,1.f,1.f,2.f,1.f,1.f,2.f,1.f,1.5f,   // 1
    1.f,1.f,1.f,1.f,1.f,2.f,1.f,1.f,1.5f,   // 2
    1.f,2.f,1.f,1.f,2.f,1.f,1.f,2.f,1.5f,   // 3
    2.f,1.f,1.f,2.f,1.f,1.f,2.f,1.f,1.5f,   // 4
    1.f,1.f,2.f,1.f,1.f,1.f,1.f,1.f,1.5f,   // 5
    1.f,2.f,1.f,1.f,2.f,1.f,1.f,2.f,1.5f,   // 6
    2.f,1.f,1.f,2.f,1.f,1.f,2.f,1.f,1.5f,   // 7
    1.5f,1.5f,1.5f,1.5f,1.5f,1.5f,1.5f,1.5f,1.5f // 8
};

__device__ __forceinline__ unsigned short f2bf(float f) {
    unsigned u = __builtin_bit_cast(unsigned, f);
    u += 0x7fffu + ((u >> 16) & 1u);               // round-to-nearest-even
    return (unsigned short)(u >> 16);
}
__device__ __forceinline__ float bf2f(unsigned short h) {
    return __builtin_bit_cast(float, ((unsigned)h) << 16);
}

// byte offset inside a 1024B fragment (32 lanes x 32B)
// A matrix 16x32 (M x K), ISA 7.12.2: lane = M + 16*((K>>3)&1),
//   VGPR0-3 hold K pairs of low 16-K half, VGPR4-7 the high half.
__device__ __forceinline__ int fragOffA(int m, int k) {
    int lane = m + (((k >> 3) & 1) << 4);
    return (lane << 5) + ((k >> 4) << 4) + (((k >> 1) & 3) << 2) + ((k & 1) << 1);
}
// B matrix 32x16 (K x N): lane = N + 16*((K>>3)&1), same K order as A.
__device__ __forceinline__ int fragOffB(int n, int k) {
    int lane = n + (((k >> 3) & 1) << 4);
    return (lane << 5) + ((k >> 4) << 4) + (((k >> 1) & 3) << 2) + ((k & 1) << 1);
}

__device__ __forceinline__ bf16x16 loadFrag(const void* p) {
    bf16x16 r;
    __builtin_memcpy(&r, p, sizeof(bf16x16));       // 32B -> 2x b128 loads
    return r;
}

// ---------------------------------------------------------------------------
// Kernel 1: convert Wq/Wk/Wv f32 (512x512, row = input k, col = output n)
// into bf16 B-fragment-linear layout in d_ws:
//   ws[w][kc 0..15][nt 0..31][512 ushort fragment]
// ---------------------------------------------------------------------------
__global__ void __launch_bounds__(THREADS)
vortex_convert_w(const float* __restrict__ Wq, const float* __restrict__ Wk,
                 const float* __restrict__ Wv, unsigned short* __restrict__ dst)
{
    int e = blockIdx.x * blockDim.x + threadIdx.x;  // 0 .. 3*262144-1
    int w = e >> 18;
    int r = e & 262143;                             // k*512 + n
    int k = r >> 9;
    int n = r & 511;
    const float* W = (w == 0) ? Wq : ((w == 1) ? Wk : Wv);
    float v = W[r];
    int frag = (k >> 5) * NTILES + (n >> 4);
    dst[(w << 18) + (frag << 9) + (fragOffB(n & 15, k & 31) >> 1)] = f2bf(v);
}

// ---------------------------------------------------------------------------
// One fully-specialized QKV projection pass:
//   C(48x512) = A(48x512) x W(512x512) in bf16 WMMA, bias add, bf16 store to LDS.
// xFrag is read-only here, all waves independent -> no barriers needed.
// ---------------------------------------------------------------------------
__device__ __forceinline__ void
qkv_gemm_pass(const char* xFrag, const unsigned short* __restrict__ wsW,
              const float* __restrict__ bias, unsigned short* dbuf,
              int lane, int wave)
{
    f32x8 acc[MTILES][4];
    #pragma unroll
    for (int mt = 0; mt < MTILES; ++mt)
        #pragma unroll
        for (int nt = 0; nt < 4; ++nt) acc[mt][nt] = (f32x8)(0.0f);

    for (int kc = 0; kc < KCHUNKS; ++kc) {
        bf16x16 a[MTILES];
        #pragma unroll
        for (int mt = 0; mt < MTILES; ++mt)
            a[mt] = loadFrag(xFrag + ((mt * KCHUNKS + kc) << 10) + (lane << 5));
        #pragma unroll
        for (int nt4 = 0; nt4 < 4; ++nt4) {
            int nt = (wave << 2) + nt4;
            bf16x16 bm = loadFrag((const char*)wsW + ((kc * NTILES + nt) << 10) + (lane << 5));
            #pragma unroll
            for (int mt = 0; mt < MTILES; ++mt)
                acc[mt][nt4] = __builtin_amdgcn_wmma_f32_16x16x32_bf16(
                    false, a[mt], false, bm, (short)0, acc[mt][nt4], false, false);
        }
    }

    // write-out: bias add, f32 -> bf16, into LDS row-major buffer.
    // Static bounds: M-tiles 0/1 rows 0..31 always live; M-tile 2 only
    // lanes 0..15 (rows 32..39) with j<4 (rows 32..35). Rows >= 36 are
    // padding and never read by the attention phase.
    #pragma unroll
    for (int nt4 = 0; nt4 < 4; ++nt4) {
        int nt  = (wave << 2) + nt4;
        int col = (nt << 4) + (lane & 15);
        float bvv = bias[col];
        #pragma unroll
        for (int mt = 0; mt < 2; ++mt) {
            int rbase = (mt << 4) + ((lane >> 4) << 3);   // C layout: VGPR j -> M=j or 8+j
            #pragma unroll
            for (int j = 0; j < 8; ++j)
                dbuf[(rbase + j) * HIDDEN + col] = f2bf(acc[mt][nt4][j] + bvv);
        }
        if (lane < 16) {
            #pragma unroll
            for (int j = 0; j < 4; ++j)
                dbuf[(32 + j) * HIDDEN + col] = f2bf(acc[2][nt4][j] + bvv);
        }
    }
}

// ---------------------------------------------------------------------------
// Kernel 2: fused QKV GEMM (WMMA bf16) + attention + gating.
// One workgroup (8 wave32) handles TB=4 batch elements.
// ---------------------------------------------------------------------------
__global__ void __launch_bounds__(THREADS, 1)
vortex_attn_kernel(const float* __restrict__ nf, const float* __restrict__ pos,
                   const unsigned short* __restrict__ wfrag,
                   const float* __restrict__ bq, const float* __restrict__ bk,
                   const float* __restrict__ bv, const float* __restrict__ Wg,
                   const float* __restrict__ bg, float* __restrict__ out, int B)
{
    extern __shared__ __align__(16) char smem[];
    char*           xFrag = smem + XFRAG_OFF;
    unsigned short* qBuf  = (unsigned short*)(smem + QBUF_OFF);
    unsigned short* kBuf  = (unsigned short*)(smem + KBUF_OFF);
    unsigned short* vBuf  = (unsigned short*)(smem + VBUF_OFF);
    float*          gates = (float*)(smem + GATES_OFF);

    const int tid  = threadIdx.x;
    const int lane = tid & 31;
    const int wave = tid >> 5;
    const int b0   = blockIdx.x * TB;
    const int maxrow = min(MROWS, (B - b0) * NUM_NODES);

    // ---- stage x = nf + pos into A-fragment layout (bf16) ------------------
    for (int e = tid; e < MROWS_PAD * HIDDEN; e += THREADS) {
        int row = e >> 9, k = e & 511;
        float v = 0.f;
        if (row < maxrow) {
            int i = row % NUM_NODES;
            v = nf[((size_t)(b0 * NUM_NODES + row)) * HIDDEN + k] + pos[i * HIDDEN + k];
        }
        int frag = (row >> 4) * KCHUNKS + (k >> 5);
        *(unsigned short*)(xFrag + (frag << 10) + fragOffA(row & 15, k & 31)) = f2bf(v);
    }
    __syncthreads();

    // ---- three fully-specialized WMMA passes (no runtime pointer selects) --
    qkv_gemm_pass(xFrag, wfrag,               bq, qBuf, lane, wave);
    qkv_gemm_pass(xFrag, wfrag + (1 << 18),   bk, kBuf, lane, wave);
    qkv_gemm_pass(xFrag, wfrag + (2 << 18),   bv, vBuf, lane, wave);

    // ---- per-batch gate: sigmoid(central . Wg + bg) ------------------------
    if (wave < TB) {
        int b = b0 + wave;
        float s = 0.f;
        if (b < B) {
            const float* cen = nf + ((size_t)(b * NUM_NODES + 8)) * HIDDEN;
            #pragma unroll
            for (int t = 0; t < 16; ++t) { int h = lane + (t << 5); s += cen[h] * Wg[h]; }
        }
        #pragma unroll
        for (int m = 16; m > 0; m >>= 1) s += __shfl_xor(s, m, 32);
        if (lane == 0) gates[wave] = 1.f / (1.f + __expf(-(s + bg[0])));
    }
    __syncthreads();

    // ---- attention: scores, softmax, weighted V, gate blend ----------------
    const float scale = 0.044194173824159216f;     // 1/sqrt(512)
    for (int r = wave; r < maxrow; r += WAVES) {
        int bb = r / NUM_NODES, i = r - bb * NUM_NODES;
        int b  = b0 + bb;

        float qv[16];
        { const unsigned short* p = qBuf + r * HIDDEN + (lane << 4);
          #pragma unroll
          for (int t = 0; t < 16; ++t) qv[t] = bf2f(p[t]); }

        float sc[NUM_NODES];
        #pragma unroll
        for (int j = 0; j < NUM_NODES; ++j) {
            const unsigned short* p = kBuf + (bb * NUM_NODES + j) * HIDDEN + (lane << 4);
            float s = 0.f;
            #pragma unroll
            for (int t = 0; t < 16; ++t) s += qv[t] * bf2f(p[t]);
            #pragma unroll
            for (int m = 16; m > 0; m >>= 1) s += __shfl_xor(s, m, 32);
            sc[j] = s;
        }
        float mmax = -1e30f;
        #pragma unroll
        for (int j = 0; j < NUM_NODES; ++j) {
            sc[j] *= scale * c_mask[i * NUM_NODES + j];
            mmax = fmaxf(mmax, sc[j]);
        }
        float ssum = 0.f;
        #pragma unroll
        for (int j = 0; j < NUM_NODES; ++j) { sc[j] = __expf(sc[j] - mmax); ssum += sc[j]; }
        float inv = 1.f / ssum;

        float ov[16];
        #pragma unroll
        for (int t = 0; t < 16; ++t) ov[t] = 0.f;
        #pragma unroll
        for (int j = 0; j < NUM_NODES; ++j) {
            float wj = sc[j] * inv;
            const unsigned short* p = vBuf + (bb * NUM_NODES + j) * HIDDEN + (lane << 4);
            #pragma unroll
            for (int t = 0; t < 16; ++t) ov[t] += wj * bf2f(p[t]);
        }
        float g = gates[bb];
        const float* cen = nf  + ((size_t)(b * NUM_NODES + 8)) * HIDDEN + (lane << 4);
        float*       op  = out + ((size_t)(b * NUM_NODES + i)) * HIDDEN + (lane << 4);
        #pragma unroll
        for (int t = 0; t < 16; ++t) op[t] = ov[t] * (1.f - g) + cen[t] * g;
    }
}

extern "C" void kernel_launch(void* const* d_in, const int* in_sizes, int n_in,
                              void* d_out, int out_size, void* d_ws, size_t ws_size,
                              hipStream_t stream) {
    const float* nf  = (const float*)d_in[0];
    const float* pos = (const float*)d_in[1];
    const float* Wq  = (const float*)d_in[2];
    const float* bq  = (const float*)d_in[3];
    const float* Wk  = (const float*)d_in[4];
    const float* bk  = (const float*)d_in[5];
    const float* Wv  = (const float*)d_in[6];
    const float* bv  = (const float*)d_in[7];
    const float* Wg  = (const float*)d_in[8];
    const float* bg  = (const float*)d_in[9];
    float* out = (float*)d_out;
    unsigned short* wfrag = (unsigned short*)d_ws;   // needs 3*512*512*2 = 1.5 MB

    int B = in_sizes[0] / (NUM_NODES * HIDDEN);

    vortex_convert_w<<<(3 * HIDDEN * HIDDEN) / THREADS, THREADS, 0, stream>>>(Wq, Wk, Wv, wfrag);

    int grid = (B + TB - 1) / TB;
    vortex_attn_kernel<<<grid, THREADS, SMEM_BYTES, stream>>>(
        nf, pos, wfrag, bq, bk, bv, Wg, bg, out, B);
}
